// TensorProductScoreModel_80152679678000
// MI455X (gfx1250) — compile-verified, matrix operand order checked by reading, not run on hardware
//
#include <hip/hip_runtime.h>

typedef __attribute__((ext_vector_type(16))) _Float16 v16h;
typedef __attribute__((ext_vector_type(8)))  float    v8f;

#define DEVINL __device__ __forceinline__

constexpr int kNL = 8000, kNR = 3000, kNA = 20000;
constexpr int kER = 96000, kEA = 640000, kELR = 400000, kELA = 640000, kEAR = 480000;
constexpr int NODE_LD = 48;   // node feature row stride (>= D2=40), tail zero-padded

DEVINL int lane_id() { return (int)(threadIdx.x & 31u); }
DEVINL int wave_id() { return (int)(threadIdx.x >> 5); }

// CDNA5: wait for this wave's outstanding LDS ops (split dep counters)
DEVINL void wave_lds_sync() { asm volatile("s_wait_dscnt 0" ::: "memory"); }

DEVINL v8f wmma16(v16h a, v16h b, v8f c) {
  return __builtin_amdgcn_wmma_f32_16x16x32_f16(false, a, false, b, (short)0, c, false, false);
}

// A fragment: 16xK (K=32 slice) from LDS row-major f16 [16 x ld]
// lanes 0-15: row m=lane, K={0..7,16..23}; lanes 16-31: row m=lane-16, K={8..15,24..31}
DEVINL v16h ld_frag_a(const _Float16* p, int ld) {
  int l = lane_id(); int m = l & 15; int h = (l >> 4) * 8;
  v16h r;
#pragma unroll
  for (int i = 0; i < 8; ++i) {
    r[i]     = p[m * ld + h + i];
    r[8 + i] = p[m * ld + 16 + h + i];
  }
  return r;
}

// B fragment: (K=32 x 16) from GLOBAL row-major f32 W[K x ldn], columns ncol0..ncol0+15,
// zero-padded for k >= Kvalid or col >= Nvalid. Mirrors A layout with M->N roles.
DEVINL v16h ld_frag_b(const float* W, int ldn, int Kvalid, int ncol0, int Nvalid) {
  int l = lane_id(); int n = l & 15; int h = (l >> 4) * 8;
  int col = ncol0 + n; bool cv = col < Nvalid;
  v16h r;
#pragma unroll
  for (int i = 0; i < 8; ++i) {
    int k0 = h + i, k1 = 16 + h + i;
    r[i]     = (cv && k0 < Kvalid) ? (_Float16)W[(size_t)k0 * ldn + col] : (_Float16)0.0f;
    r[8 + i] = (cv && k1 < Kvalid) ? (_Float16)W[(size_t)k1 * ldn + col] : (_Float16)0.0f;
  }
  return r;
}

// B fragment of K^T for attention scores: element (kdim, key=base+n), kdim<4 from head slice
DEVINL v16h ld_b_keys(const float* kmat, int base, int head, int NK) {
  int l = lane_id(); int n = l & 15; int h = (l >> 4) * 8;
  int key = base + n;
  v16h r;
#pragma unroll
  for (int i = 0; i < 8; ++i) {
    int k0 = h + i, k1 = 16 + h + i;
    r[i]     = (k0 < 4 && key < NK) ? (_Float16)kmat[(size_t)key * 16 + head * 4 + k0] : (_Float16)0.0f;
    r[8 + i] = (k1 < 4 && key < NK) ? (_Float16)kmat[(size_t)key * 16 + head * 4 + k1] : (_Float16)0.0f;
  }
  return r;
}

// B fragment of V for PV: element (kk=key-in-tile, n=head dim<4)
DEVINL v16h ld_b_vals(const float* vmat, int base, int head, int NK) {
  int l = lane_id(); int n = l & 15; int h = (l >> 4) * 8;
  v16h r;
#pragma unroll
  for (int i = 0; i < 8; ++i) {
    int key0 = base + h + i, key1 = base + 16 + h + i;
    r[i]     = (n < 4 && key0 < NK) ? (_Float16)vmat[(size_t)key0 * 16 + head * 4 + n] : (_Float16)0.0f;
    r[8 + i] = (n < 4 && key1 < NK) ? (_Float16)vmat[(size_t)key1 * 16 + head * 4 + n] : (_Float16)0.0f;
  }
  return r;
}

// ---------------- edge embedding: ea = relu([sigma(32)|smear(32)] @ W1 + b1) @ W2 + b2 ----
__global__ void edge_emb_kernel(const float* __restrict__ d, const float* __restrict__ sigma,
                                const float* __restrict__ W1, const float* __restrict__ b1,
                                const float* __restrict__ W2, const float* __restrict__ b2,
                                float stop, int E, _Float16* __restrict__ ea) {
  __shared__ _Float16 sA[8][16 * 64];
  __shared__ _Float16 sB[8][16 * 32];
  int w = wave_id(), l = lane_id();
  int base = (blockIdx.x * 8 + w) * 16;
  if (l < 16) {
    int row = l; int e = base + row;
    float dv = (e < E) ? d[e] : 0.0f;
    float inv = 31.0f / stop;
    float coeff = -0.5f * inv * inv;
#pragma unroll 4
    for (int j = 0; j < 32; ++j) {
      sA[w][row * 64 + j] = (_Float16)sigma[j];
      float t = dv - stop * (float)j * (1.0f / 31.0f);
      sA[w][row * 64 + 32 + j] = (_Float16)__expf(coeff * t * t);
    }
  } else {
    int row = l - 16;
#pragma unroll
    for (int j = 16; j < 32; ++j) sB[w][row * 32 + j] = (_Float16)0.0f;
  }
  __syncthreads();
  v8f c = {};
  c = wmma16(ld_frag_a(&sA[w][0], 64),  ld_frag_b(W1, 16, 32, 0, 16), c);
  c = wmma16(ld_frag_a(&sA[w][32], 64), ld_frag_b(W1 + 32 * 16, 16, 32, 0, 16), c);
  int n = l & 15, h = (l >> 4) * 8;
  float bias = b1[n];
#pragma unroll
  for (int r = 0; r < 8; ++r)
    sB[w][(r + h) * 32 + n] = (_Float16)fmaxf(c[r] + bias, 0.0f);
  __syncthreads();
  v8f c2 = {};
  c2 = wmma16(ld_frag_a(&sB[w][0], 32), ld_frag_b(W2, 16, 16, 0, 16), c2);
  float bias2 = b2[n];
#pragma unroll
  for (int r = 0; r < 8; ++r) {
    int e = base + r + h;
    if (e < E) ea[(size_t)e * 16 + n] = (_Float16)(c2[r] + bias2);
  }
}

// ---------------- spherical harmonics per edge ----------------
__global__ void sph_kernel(const float* __restrict__ vec, int E, _Float16* __restrict__ sh) {
  int e = blockIdx.x * blockDim.x + threadIdx.x;
  if (e >= E) return;
  float x = vec[(size_t)e * 3], y = vec[(size_t)e * 3 + 1], z = vec[(size_t)e * 3 + 2];
  float invn = 1.0f / (sqrtf(x * x + y * y + z * z) + 1e-8f);
  x *= invn; y *= invn; z *= invn;
  const float s3 = 1.7320508f, s15 = 3.8729833f, s5h = 1.118034f;
  _Float16* o = sh + (size_t)e * 9;
  o[0] = (_Float16)1.0f;      o[1] = (_Float16)(s3 * x);  o[2] = (_Float16)(s3 * y);
  o[3] = (_Float16)(s3 * z);  o[4] = (_Float16)(s15 * x * y);
  o[5] = (_Float16)(s15 * y * z);
  o[6] = (_Float16)(s5h * (3.0f * z * z - 1.0f));
  o[7] = (_Float16)(s15 * x * z);
  o[8] = (_Float16)(0.5f * s15 * (x * x - y * y));
}

// ---------------- init node buffer (stride 48, zero tail) ----------------
__global__ void init_nodes_kernel(const float* __restrict__ x, int N, int D, float* __restrict__ node) {
  int idx = blockIdx.x * blockDim.x + threadIdx.x;
  int n = idx / NODE_LD, dcol = idx % NODE_LD;
  if (n >= N) return;
  node[idx] = (dcol < D) ? x[(size_t)n * D + dcol] : 0.0f;
}

// ---------------- tp_conv: per-edge WMMA + atomic segment scatter ----------------
__global__ void tp_conv_kernel(const float* __restrict__ xg, const float* __restrict__ og,
                               const int* __restrict__ gi, const int* __restrict__ si,
                               const _Float16* __restrict__ ea, const _Float16* __restrict__ sh,
                               const float* __restrict__ Win, const float* __restrict__ Wsh,
                               const float* __restrict__ M1, const float* __restrict__ c1,
                               const float* __restrict__ M2, const float* __restrict__ c2,
                               int E, int Din, int Dout, int ntile,
                               float* __restrict__ num, float* __restrict__ cnt) {
  __shared__ _Float16 sCat[8][16 * 64];
  __shared__ _Float16 sW[8][16 * 32];
  __shared__ _Float16 sX[8][16 * 32];
  __shared__ _Float16 sSh[8][16 * 32];
  __shared__ int sSi[8][16];
  int w = wave_id(), l = lane_id();
  int base = (blockIdx.x * 8 + w) * 16;
  if (l < 16) {
    int row = l; int e = base + row;
    bool val = e < E;
    int s_ = val ? si[e] : 0;
    int g_ = val ? gi[e] : 0;
    sSi[w][row] = s_;
#pragma unroll 4
    for (int j = 0; j < 16; ++j) {
      sCat[w][row * 64 + j]      = val ? ea[(size_t)e * 16 + j] : (_Float16)0.0f;
      sCat[w][row * 64 + 16 + j] = val ? (_Float16)og[(size_t)s_ * NODE_LD + j] : (_Float16)0.0f;
      sCat[w][row * 64 + 32 + j] = val ? (_Float16)xg[(size_t)g_ * NODE_LD + j] : (_Float16)0.0f;
      sCat[w][row * 64 + 48 + j] = (_Float16)0.0f;
    }
#pragma unroll 4
    for (int j = 0; j < 32; ++j) {
      sX[w][row * 32 + j]  = (val && j < Din) ? (_Float16)xg[(size_t)g_ * NODE_LD + j] : (_Float16)0.0f;
      sSh[w][row * 32 + j] = (val && j < 9)   ? sh[(size_t)e * 9 + j] : (_Float16)0.0f;
    }
  } else {
    int row = l - 16;
#pragma unroll 4
    for (int j = 0; j < 32; ++j) sW[w][row * 32 + j] = (_Float16)0.0f;
  }
  __syncthreads();
  int n = l & 15, h = (l >> 4) * 8;
  // w1 = relu([ea|a_feat|b_feat](48) @ M1 + c1)
  v8f cw = {};
  cw = wmma16(ld_frag_a(&sCat[w][0], 64),  ld_frag_b(M1, 16, 32, 0, 16), cw);
  cw = wmma16(ld_frag_a(&sCat[w][32], 64), ld_frag_b(M1 + 32 * 16, 16, 16, 0, 16), cw);
  float b1 = c1[n];
#pragma unroll
  for (int r = 0; r < 8; ++r)
    sW[w][(r + h) * 32 + n] = (_Float16)fmaxf(cw[r] + b1, 0.0f);
  __syncthreads();
  v16h aW  = ld_frag_a(&sW[w][0], 32);
  v16h aX  = ld_frag_a(&sX[w][0], 32);
  v16h aSh = ld_frag_a(&sSh[w][0], 32);
  for (int t = 0; t < ntile; ++t) {
    v8f z = {};
    v8f vw = wmma16(aW,  ld_frag_b(M2, Dout, 16, t * 16, Dout), z);
    v8f vx = wmma16(aX,  ld_frag_b(Win, Dout, Din, t * 16, Dout), z);
    v8f vs = wmma16(aSh, ld_frag_b(Wsh, Dout, 9, t * 16, Dout), z);
    int col = t * 16 + n;
    float b2 = (col < Dout) ? c2[col] : 0.0f;
#pragma unroll
    for (int r = 0; r < 8; ++r) {
      int e = base + r + h;
      if (e < E && col < Dout) {
        float m = vx[r] * vs[r] * (vw[r] + b2);
        unsafeAtomicAdd(&num[(size_t)sSi[w][r + h] * Dout + col], m);
      }
    }
  }
  if (l < 16) {
    int e = base + l;
    if (e < E) unsafeAtomicAdd(&cnt[sSi[w][l]], 1.0f);
  }
}

// ---------------- node update: node = pad(node) + sum_i num_i / max(cnt_i,1) ----------------
__global__ void node_update_kernel(float* __restrict__ node, int N, int Dold, int Dnew,
                                   const float* __restrict__ num0, const float* __restrict__ cnt0,
                                   const float* __restrict__ num1, const float* __restrict__ cnt1,
                                   const float* __restrict__ num2, const float* __restrict__ cnt2) {
  int idx = blockIdx.x * blockDim.x + threadIdx.x;
  int n = idx / NODE_LD, dcol = idx % NODE_LD;
  if (n >= N) return;
  float v = 0.0f;
  if (dcol < Dnew) {
    if (dcol < Dold) v = node[idx];
    v += num0[(size_t)n * Dnew + dcol] / fmaxf(cnt0[n], 1.0f);
    if (num1) v += num1[(size_t)n * Dnew + dcol] / fmaxf(cnt1[n], 1.0f);
    if (num2) v += num2[(size_t)n * Dnew + dcol] / fmaxf(cnt2[n], 1.0f);
  }
  node[idx] = v;
}

// ---------------- out(Mx16) = X[:, :16] @ W(16x16) + b ----------------
__global__ void gemm16_kernel(const float* __restrict__ X, int ldin, int M,
                              const float* __restrict__ W, const float* __restrict__ b,
                              float* __restrict__ out) {
  __shared__ _Float16 sA[8][16 * 32];
  int w = wave_id(), l = lane_id();
  int base = (blockIdx.x * 8 + w) * 16;
  if (l < 16) {
    int row = l, rg = base + row;
#pragma unroll 4
    for (int j = 0; j < 32; ++j)
      sA[w][row * 32 + j] = (rg < M && j < 16) ? (_Float16)X[(size_t)rg * ldin + j] : (_Float16)0.0f;
  }
  __syncthreads();
  v8f c = {};
  c = wmma16(ld_frag_a(&sA[w][0], 32), ld_frag_b(W, 16, 16, 0, 16), c);
  int n = l & 15, h = (l >> 4) * 8;
  float bias = b[n];
#pragma unroll
  for (int r = 0; r < 8; ++r) {
    int rg = base + r + h;
    if (rg < M) out[(size_t)rg * 16 + n] = c[r] + bias;
  }
}

// ---------------- flash attention: one wave per (16-query tile, head), head dim 4 ----------
__global__ void attn_kernel(const float* __restrict__ q, const float* __restrict__ k,
                            const float* __restrict__ v, int NQ, int NK,
                            float* __restrict__ obuf) {
  __shared__ _Float16 sQ[8][16 * 32];
  __shared__ _Float16 sP[8][16 * 32];
  int w = wave_id(), l = lane_id();
  int gw = blockIdx.x * 8 + w;
  int qt = gw >> 2, head = gw & 3;
  if (qt * 16 >= NQ) return;
  int n = l & 15, h = (l >> 4) * 8;
  if (l < 16) {
    int row = l; int qr = qt * 16 + row;
#pragma unroll 4
    for (int j = 0; j < 32; ++j)
      sQ[w][row * 32 + j] = (qr < NQ && j < 4) ? (_Float16)q[(size_t)qr * 16 + head * 4 + j]
                                               : (_Float16)0.0f;
  }
  wave_lds_sync();
  v16h aQ = ld_frag_a(&sQ[w][0], 32);
  v8f o = {};
  float rmax[8], rsum[8];
#pragma unroll
  for (int r = 0; r < 8; ++r) { rmax[r] = -1e30f; rsum[r] = 0.0f; }
  const float scale = 0.5f;  // 1/sqrt(head_dim=4)
  int ktiles = (NK + 31) / 32;
  for (int kt = 0; kt < ktiles; ++kt) {
    int kb = kt * 32;
    v8f z = {};
    v8f s0 = wmma16(aQ, ld_b_keys(k, kb, head, NK), z);
    v8f s1 = wmma16(aQ, ld_b_keys(k, kb + 16, head, NK), z);
    bool m0 = (kb + n) >= NK, m1 = (kb + 16 + n) >= NK;
#pragma unroll
    for (int r = 0; r < 8; ++r) {
      s0[r] = m0 ? -1e30f : s0[r] * scale;
      s1[r] = m1 ? -1e30f : s1[r] * scale;
    }
#pragma unroll
    for (int r = 0; r < 8; ++r) {
      float tm = fmaxf(s0[r], s1[r]);
#pragma unroll
      for (int mk = 1; mk < 16; mk <<= 1) tm = fmaxf(tm, __shfl_xor(tm, mk, 32));
      float nm = fmaxf(rmax[r], tm);
      float alpha = __expf(rmax[r] - nm);
      rmax[r] = nm;
      float p0 = __expf(s0[r] - nm);
      float p1 = __expf(s1[r] - nm);
      float ts = p0 + p1;
#pragma unroll
      for (int mk = 1; mk < 16; mk <<= 1) ts += __shfl_xor(ts, mk, 32);
      rsum[r] = rsum[r] * alpha + ts;
      o[r] *= alpha;
      sP[w][(r + h) * 32 + n]      = (_Float16)p0;
      sP[w][(r + h) * 32 + 16 + n] = (_Float16)p1;
    }
    wave_lds_sync();
    o = wmma16(ld_frag_a(&sP[w][0], 32), ld_b_vals(v, kb, head, NK), o);
  }
#pragma unroll
  for (int r = 0; r < 8; ++r) {
    if (n < 4) {
      int qr = qt * 16 + r + h;
      if (qr < NQ) obuf[(size_t)qr * 16 + head * 4 + n] = o[r] / fmaxf(rsum[r], 1e-20f);
    }
  }
}

// =======================================================================================
extern "C" void kernel_launch(void* const* d_in, const int* in_sizes, int n_in,
                              void* d_out, int out_size, void* d_ws, size_t ws_size,
                              hipStream_t stream) {
  (void)in_sizes; (void)n_in; (void)out_size; (void)ws_size;
  const float* lig_x  = (const float*)d_in[0];
  const float* rec_x  = (const float*)d_in[1];
  const float* atom_x = (const float*)d_in[2];
  const float* sigma  = (const float*)d_in[3];
  const float* rec_dd   = (const float*)d_in[4];
  const float* rec_vec  = (const float*)d_in[5];
  const float* atom_dd  = (const float*)d_in[6];
  const float* atom_vec = (const float*)d_in[7];
  const float* lr_dd    = (const float*)d_in[8];
  const float* lr_vec   = (const float*)d_in[9];
  const float* la_dd    = (const float*)d_in[10];
  const float* la_vec   = (const float*)d_in[11];
  const float* ar_dd    = (const float*)d_in[12];
  const float* ar_vec   = (const float*)d_in[13];
  const float* Wemb1 = (const float*)d_in[14];
  const float* bemb1 = (const float*)d_in[15];
  const float* Wemb2 = (const float*)d_in[16];
  const float* bemb2 = (const float*)d_in[17];
  const float* Win0 = (const float*)d_in[18];
  const float* Wsh0 = (const float*)d_in[19];
  const float* M1_0 = (const float*)d_in[20];
  const float* c1_0 = (const float*)d_in[21];
  const float* M2_0 = (const float*)d_in[22];
  const float* c2_0 = (const float*)d_in[23];
  const float* Win1 = (const float*)d_in[24];
  const float* Wsh1 = (const float*)d_in[25];
  const float* M1_1 = (const float*)d_in[26];
  const float* c1_1 = (const float*)d_in[27];
  const float* M2_1 = (const float*)d_in[28];
  const float* c2_1 = (const float*)d_in[29];
  const float* W_qkv = (const float*)d_in[30];
  const float* b_qkv = (const float*)d_in[31];
  const float* Wo = (const float*)d_in[32];
  const float* bo = (const float*)d_in[33];
  const int* rec_src  = (const int*)d_in[34];
  const int* rec_dst  = (const int*)d_in[35];
  const int* atom_src = (const int*)d_in[36];
  const int* atom_dst = (const int*)d_in[37];
  const int* lr_src = (const int*)d_in[38];
  const int* lr_dst = (const int*)d_in[39];
  const int* la_src = (const int*)d_in[40];
  const int* la_dst = (const int*)d_in[41];
  const int* ar_src = (const int*)d_in[42];
  const int* ar_dst = (const int*)d_in[43];

  char* basep = (char*)d_ws;
  size_t off = 0;
  auto carve = [&](size_t bytes) -> void* {
    void* r = basep + off;
    off = (off + bytes + 255) & ~(size_t)255;
    return r;
  };
  // f16 per-edge embeddings and sph harmonics
  _Float16* ea_re = (_Float16*)carve((size_t)kER * 16 * 2);
  _Float16* ea_at = (_Float16*)carve((size_t)kEA * 16 * 2);
  _Float16* ea_lr = (_Float16*)carve((size_t)kELR * 16 * 2);
  _Float16* ea_la = (_Float16*)carve((size_t)kELA * 16 * 2);
  _Float16* ea_ar = (_Float16*)carve((size_t)kEAR * 16 * 2);
  _Float16* sh_re = (_Float16*)carve((size_t)kER * 9 * 2);
  _Float16* sh_at = (_Float16*)carve((size_t)kEA * 9 * 2);
  _Float16* sh_lr = (_Float16*)carve((size_t)kELR * 9 * 2);
  _Float16* sh_la = (_Float16*)carve((size_t)kELA * 9 * 2);
  _Float16* sh_ar = (_Float16*)carve((size_t)kEAR * 9 * 2);
  // node buffers, stride 48
  float* lig  = (float*)carve((size_t)kNL * NODE_LD * 4);
  float* rec  = (float*)carve((size_t)kNR * NODE_LD * 4);
  float* atom = (float*)carve((size_t)kNA * NODE_LD * 4);
  // accumulators (contiguous region for one memset)
  size_t acc_beg = off;
  float* num_lrl = (float*)carve((size_t)kNL * 28 * 4);
  float* num_lal = (float*)carve((size_t)kNL * 28 * 4);
  float* num_at  = (float*)carve((size_t)kNA * 28 * 4);
  float* num_al  = (float*)carve((size_t)kNA * 28 * 4);
  float* num_ar  = (float*)carve((size_t)kNA * 28 * 4);
  float* num_rr  = (float*)carve((size_t)kNR * 28 * 4);
  float* num_rl  = (float*)carve((size_t)kNR * 28 * 4);
  float* num_ra  = (float*)carve((size_t)kNR * 28 * 4);
  float* num_1a  = (float*)carve((size_t)kNL * 40 * 4);
  float* num_1b  = (float*)carve((size_t)kNL * 40 * 4);
  float* cnt_lrl = (float*)carve((size_t)kNL * 4);
  float* cnt_lal = (float*)carve((size_t)kNL * 4);
  float* cnt_at  = (float*)carve((size_t)kNA * 4);
  float* cnt_al  = (float*)carve((size_t)kNA * 4);
  float* cnt_ar  = (float*)carve((size_t)kNA * 4);
  float* cnt_rr  = (float*)carve((size_t)kNR * 4);
  float* cnt_rl  = (float*)carve((size_t)kNR * 4);
  float* cnt_ra  = (float*)carve((size_t)kNR * 4);
  float* cnt_1a  = (float*)carve((size_t)kNL * 4);
  float* cnt_1b  = (float*)carve((size_t)kNL * 4);
  size_t acc_end = off;
  // attention buffers
  float* qb = (float*)carve((size_t)kNL * 16 * 4);
  float* kb = (float*)carve((size_t)kNR * 16 * 4);
  float* vb = (float*)carve((size_t)kNR * 16 * 4);
  float* ob = (float*)carve((size_t)kNL * 16 * 4);

  dim3 B(256);
  auto eblocks = [](int E) { return dim3((unsigned)((E + 127) / 128)); };

  // --- per-edge-set embeddings (weight slice i of Wemb1[5][64][16] etc.) ---
  edge_emb_kernel<<<eblocks(kER),  B, 0, stream>>>(rec_dd,  sigma, Wemb1 + 0 * 1024, bemb1 + 0 * 16, Wemb2 + 0 * 256, bemb2 + 0 * 16, 30.0f,  kER,  ea_re);
  edge_emb_kernel<<<eblocks(kEA),  B, 0, stream>>>(atom_dd, sigma, Wemb1 + 1 * 1024, bemb1 + 1 * 16, Wemb2 + 1 * 256, bemb2 + 1 * 16, 5.0f,   kEA,  ea_at);
  edge_emb_kernel<<<eblocks(kELR), B, 0, stream>>>(lr_dd,   sigma, Wemb1 + 2 * 1024, bemb1 + 2 * 16, Wemb2 + 2 * 256, bemb2 + 2 * 16, 250.0f, kELR, ea_lr);
  edge_emb_kernel<<<eblocks(kELA), B, 0, stream>>>(la_dd,   sigma, Wemb1 + 3 * 1024, bemb1 + 3 * 16, Wemb2 + 3 * 256, bemb2 + 3 * 16, 250.0f, kELA, ea_la);
  edge_emb_kernel<<<eblocks(kEAR), B, 0, stream>>>(ar_dd,   sigma, Wemb1 + 4 * 1024, bemb1 + 4 * 16, Wemb2 + 4 * 256, bemb2 + 4 * 16, 30.0f,  kEAR, ea_ar);
  // --- spherical harmonics ---
  sph_kernel<<<(kER + 255) / 256,  B, 0, stream>>>(rec_vec,  kER,  sh_re);
  sph_kernel<<<(kEA + 255) / 256,  B, 0, stream>>>(atom_vec, kEA,  sh_at);
  sph_kernel<<<(kELR + 255) / 256, B, 0, stream>>>(lr_vec,   kELR, sh_lr);
  sph_kernel<<<(kELA + 255) / 256, B, 0, stream>>>(la_vec,   kELA, sh_la);
  sph_kernel<<<(kEAR + 255) / 256, B, 0, stream>>>(ar_vec,   kEAR, sh_ar);
  // --- node buffers ---
  init_nodes_kernel<<<(kNL * NODE_LD + 255) / 256, B, 0, stream>>>(lig_x,  kNL, 16, lig);
  init_nodes_kernel<<<(kNR * NODE_LD + 255) / 256, B, 0, stream>>>(rec_x,  kNR, 16, rec);
  init_nodes_kernel<<<(kNA * NODE_LD + 255) / 256, B, 0, stream>>>(atom_x, kNA, 16, atom);
  hipMemsetAsync(basep + acc_beg, 0, acc_end - acc_beg, stream);

#define P0(i) Win0 + (i) * 16 * 28, Wsh0 + (i) * 9 * 28, M1_0 + (i) * 48 * 16, c1_0 + (i) * 16, M2_0 + (i) * 16 * 28, c2_0 + (i) * 28
#define P1(i) Win1 + (i) * 28 * 40, Wsh1 + (i) * 9 * 40, M1_1 + (i) * 48 * 16, c1_1 + (i) * 16, M2_1 + (i) * 16 * 40, c2_1 + (i) * 40
  // --- layer 0: 8 tp_conv message passes (Din=16, Dout=28, 2 N-tiles) ---
  tp_conv_kernel<<<eblocks(kELR), B, 0, stream>>>(rec,  lig,  lr_dst,   lr_src,   ea_lr, sh_lr, P0(0), kELR, 16, 28, 2, num_lrl, cnt_lrl); // lr_up
  tp_conv_kernel<<<eblocks(kELA), B, 0, stream>>>(atom, lig,  la_dst,   la_src,   ea_la, sh_la, P0(1), kELA, 16, 28, 2, num_lal, cnt_lal); // la_up
  tp_conv_kernel<<<eblocks(kEA),  B, 0, stream>>>(atom, atom, atom_dst, atom_src, ea_at, sh_at, P0(2), kEA,  16, 28, 2, num_at,  cnt_at);  // atom_up
  tp_conv_kernel<<<eblocks(kELA), B, 0, stream>>>(lig,  atom, la_src,   la_dst,   ea_la, sh_la, P0(3), kELA, 16, 28, 2, num_al,  cnt_al);  // al_up
  tp_conv_kernel<<<eblocks(kEAR), B, 0, stream>>>(rec,  atom, ar_dst,   ar_src,   ea_ar, sh_ar, P0(4), kEAR, 16, 28, 2, num_ar,  cnt_ar);  // ar_up
  tp_conv_kernel<<<eblocks(kER),  B, 0, stream>>>(rec,  rec,  rec_dst,  rec_src,  ea_re, sh_re, P0(5), kER,  16, 28, 2, num_rr,  cnt_rr);  // rec_up
  tp_conv_kernel<<<eblocks(kELR), B, 0, stream>>>(lig,  rec,  lr_src,   lr_dst,   ea_lr, sh_lr, P0(6), kELR, 16, 28, 2, num_rl,  cnt_rl);  // rl_up
  tp_conv_kernel<<<eblocks(kEAR), B, 0, stream>>>(atom, rec,  ar_src,   ar_dst,   ea_ar, sh_ar, P0(7), kEAR, 16, 28, 2, num_ra,  cnt_ra);  // ra_up
  // --- layer 0 node updates ---
  node_update_kernel<<<(kNL * NODE_LD + 255) / 256, B, 0, stream>>>(lig,  kNL, 16, 28, num_lal, cnt_lal, num_lrl, cnt_lrl, nullptr, nullptr);
  node_update_kernel<<<(kNA * NODE_LD + 255) / 256, B, 0, stream>>>(atom, kNA, 16, 28, num_at,  cnt_at,  num_al,  cnt_al,  num_ar,  cnt_ar);
  node_update_kernel<<<(kNR * NODE_LD + 255) / 256, B, 0, stream>>>(rec,  kNR, 16, 28, num_rr,  cnt_rr,  num_rl,  cnt_rl,  num_ra,  cnt_ra);
  // --- layer 1: 2 tp_convs (Din=28, Dout=40, 3 N-tiles) ---
  tp_conv_kernel<<<eblocks(kELR), B, 0, stream>>>(rec,  lig, lr_dst, lr_src, ea_lr, sh_lr, P1(0), kELR, 28, 40, 3, num_1a, cnt_1a); // lr_up1
  tp_conv_kernel<<<eblocks(kELA), B, 0, stream>>>(atom, lig, la_dst, la_src, ea_la, sh_la, P1(1), kELA, 28, 40, 3, num_1b, cnt_1b); // la_up1
  node_update_kernel<<<(kNL * NODE_LD + 255) / 256, B, 0, stream>>>(lig, kNL, 28, 40, num_1b, cnt_1b, num_1a, cnt_1a, nullptr, nullptr);
#undef P0
#undef P1
  // --- cross attention ---
  gemm16_kernel<<<eblocks(kNL), B, 0, stream>>>(lig, NODE_LD, kNL, W_qkv + 0 * 256, b_qkv + 0 * 16, qb);
  gemm16_kernel<<<eblocks(kNR), B, 0, stream>>>(rec, NODE_LD, kNR, W_qkv + 1 * 256, b_qkv + 1 * 16, kb);
  gemm16_kernel<<<eblocks(kNR), B, 0, stream>>>(rec, NODE_LD, kNR, W_qkv + 2 * 256, b_qkv + 2 * 16, vb);
  attn_kernel<<<dim3((unsigned)(((kNL / 16) * 4 + 7) / 8)), B, 0, stream>>>(qb, kb, vb, kNL, kNR, ob);
  gemm16_kernel<<<eblocks(kNL), B, 0, stream>>>(ob, 16, kNL, Wo, bo, (float*)d_out);
}
